// GINAffinityNet_16037407883662
// MI455X (gfx1250) — matrix operand order repaired and checked by reading, not compiled, not run on hardware
//
#include <hip/hip_runtime.h>

// ---------------------------------------------------------------------------
// Problem constants (match reference)
// ---------------------------------------------------------------------------
#define CN 200000
#define CF 64
#define CH 256
#define CG 4096
#define CT 1280
#define CE 800000
#define BN_EPS 1e-5f

typedef __attribute__((ext_vector_type(16))) __bf16 bf16x16;
typedef __attribute__((ext_vector_type(8)))  __bf16 bf16x8;
typedef __attribute__((ext_vector_type(4)))  __bf16 bf16x4;
typedef __attribute__((ext_vector_type(8)))  float  f32x8;
typedef __attribute__((ext_vector_type(4)))  float  f32x4;

// ---------------------------------------------------------------------------
// Small utility kernels
// ---------------------------------------------------------------------------
__global__ void k_zero4(f32x4* __restrict__ p, long n4) {
  long i = (long)blockIdx.x * blockDim.x + threadIdx.x;
  if (i < n4) p[i] = (f32x4)0.f;
}

__global__ void k_f32_to_bf16(const float* __restrict__ s, __bf16* __restrict__ d, int n) {
  int i = blockIdx.x * blockDim.x + threadIdx.x;
  if (i < n) d[i] = (__bf16)s[i];
}

// agg[dst[e]] += h[src[e]]  (H floats per edge, float4 + 4 atomics per thread)
__global__ void k_edge_scatter(const float* __restrict__ h,
                               const int* __restrict__ src,
                               const int* __restrict__ dst,
                               float* __restrict__ agg,
                               int E, int H, int shift) {
  long gid = (long)blockIdx.x * blockDim.x + threadIdx.x;
  long total = (long)E << shift;          // E * (H/4)
  if (gid >= total) return;
  int e = (int)(gid >> shift);
  int c = (int)((gid & ((1 << shift) - 1)) << 2);
  int s = src[e], d = dst[e];
  f32x4 v = *(const f32x4*)(h + (long)s * H + c);
  float* o = agg + (long)d * H + c;
  atomicAdd(o + 0, v[0]);
  atomicAdd(o + 1, v[1]);
  atomicAdd(o + 2, v[2]);
  atomicAdd(o + 3, v[3]);
}

// combined[batch[n], 0:256] += h[n, :]   (ld of combined = 512)
__global__ void k_pool(const float* __restrict__ h,
                       const int* __restrict__ batch,
                       float* __restrict__ comb, int N) {
  long gid = (long)blockIdx.x * blockDim.x + threadIdx.x;
  long total = (long)N * 64;              // N * 256/4
  if (gid >= total) return;
  int n = (int)(gid >> 6);
  int c = (int)((gid & 63) << 2);
  int g = batch[n];
  f32x4 v = *(const f32x4*)(h + (long)n * CH + c);
  float* o = comb + (long)g * 512 + c;
  atomicAdd(o + 0, v[0]);
  atomicAdd(o + 1, v[1]);
  atomicAdd(o + 2, v[2]);
  atomicAdd(o + 3, v[3]);
}

// Per-column sum / sum-of-squares over z[N,256]
__global__ void k_bn_reduce(const float* __restrict__ z,
                            float* __restrict__ csum,
                            float* __restrict__ csq, int N) {
  int col = threadIdx.x;                  // 256 threads = 256 columns
  int row0 = blockIdx.x * 512;
  float s = 0.f, s2 = 0.f;
  for (int r = 0; r < 512; ++r) {
    int row = row0 + r;
    if (row < N) {
      float v = z[(long)row * CH + col];
      s += v; s2 += v * v;
    }
  }
  atomicAdd(&csum[col], s);
  atomicAdd(&csq[col], s2);
}

// h = relu((z - mu) * rsqrt(var+eps) * gamma + beta)  -- float4 per thread
__global__ void k_bn_apply(const float* __restrict__ z,
                           const float* __restrict__ csum,
                           const float* __restrict__ csq,
                           const float* __restrict__ gamma,
                           const float* __restrict__ beta,
                           float* __restrict__ out, int N) {
  long i4 = (long)blockIdx.x * blockDim.x + threadIdx.x;
  long total4 = (long)N * (CH / 4);
  if (i4 >= total4) return;
  int c0 = (int)((i4 << 2) & (CH - 1));
  const float inv_n = 1.0f / (float)N;
  f32x4 v = *(const f32x4*)(z + (i4 << 2));
  f32x4 r;
#pragma unroll
  for (int q = 0; q < 4; ++q) {
    int c = c0 + q;
    float mu = csum[c] * inv_n;
    float var = csq[c] * inv_n - mu * mu;
    float t = (v[q] - mu) * rsqrtf(var + BN_EPS) * gamma[c] + beta[c];
    r[q] = fmaxf(t, 0.f);
  }
  *(f32x4*)(out + (i4 << 2)) = r;
}

// out[g] = r1[g,:] . w + b   (wave32 per row)
__global__ void k_gemv_out(const float* __restrict__ r1,
                           const float* __restrict__ w,
                           const float* __restrict__ b,
                           float* __restrict__ out, int G) {
  int wid = threadIdx.x >> 5;
  int lane = threadIdx.x & 31;
  int row = blockIdx.x * 8 + wid;
  if (row >= G) return;
  float s = 0.f;
#pragma unroll
  for (int t = 0; t < 8; ++t) {
    int c = lane + 32 * t;
    s += r1[(long)row * CH + c] * w[c];
  }
#pragma unroll
  for (int off = 16; off > 0; off >>= 1) s += __shfl_down(s, off);
  if (lane == 0) out[row] = s + b[0];
}

// ---------------------------------------------------------------------------
// WMMA GEMM:  C[M,256] = act( (A1 (+A2)) [M,K] x W[K,256] + bias )
// bf16 inputs, fp32 accumulate with v_wmma_f32_16x16x32_bf16.
// Block = 256 threads (8 wave32 waves), tile = 128x128, wave tile = 32x64.
// Double-buffered LDS; B (weight) tiles staged with the CDNA5 async-to-LDS
// engine (global_load_async_to_lds_b128, ASYNCcnt), A tiles staged through
// VGPRs with fused A1+A2 add and fp32->bf16 conversion.
// ---------------------------------------------------------------------------
#define AP 40    // A LDS pitch in bf16 (32 + 8 pad)
#define BP 136   // B LDS pitch in bf16 (128 + 8 pad)

template <bool HAS_A2>
__global__ __launch_bounds__(256) void k_wmma_gemm(
    const float* __restrict__ A1, const float* __restrict__ A2,
    const __bf16* __restrict__ W, const float* __restrict__ bias,
    float* __restrict__ C, int M, int K, int ldc, int relu) {
  __shared__ __bf16 Alds[2][128 * AP];
  __shared__ __bf16 Blds[2][32 * BP];

  const int tid = threadIdx.x;
  const int wid = tid >> 5;
  const int lane = tid & 31;
  const int wm = wid >> 1;          // 0..3  (row waves)
  const int wn = wid & 1;           // 0..1  (col waves)
  const int rowBlock = blockIdx.x * 128;
  const int colBlock = blockIdx.y * 128;

  f32x8 acc[2][4];
#pragma unroll
  for (int i = 0; i < 2; ++i)
#pragma unroll
    for (int j = 0; j < 4; ++j) acc[i][j] = (f32x8)0.f;

  // --- per-thread staging coordinates (straight-line, no divergence) ---
  int arow[4], akc[4];
  long aoff[4];
#pragma unroll
  for (int it = 0; it < 4; ++it) {
    int slot = tid + it * 256;        // 0..1023
    int r = slot >> 3;                // 0..127
    int kc = (slot & 7) << 2;         // 0,4,..,28
    int grow = rowBlock + r;
    grow = grow < M ? grow : (M - 1); // clamp: clamped rows never stored
    arow[it] = r; akc[it] = kc;
    aoff[it] = (long)grow * K + kc;
  }
  int brow[2], bcc[2];
#pragma unroll
  for (int it = 0; it < 2; ++it) {
    int slot = tid + it * 256;        // 0..511
    brow[it] = slot >> 4;             // 0..31
    bcc[it] = (slot & 15) << 3;       // 0,8,..,120
  }

  f32x4 aReg[4];

  auto loadA = [&](int kt) {
#pragma unroll
    for (int it = 0; it < 4; ++it) {
      f32x4 v = *(const f32x4*)(A1 + aoff[it] + kt);
      if (HAS_A2) {
        f32x4 u = *(const f32x4*)(A2 + aoff[it] + kt);
        v += u;
      }
      aReg[it] = v;
    }
  };
  auto storeA = [&](int buf) {
#pragma unroll
    for (int it = 0; it < 4; ++it) {
      bf16x4 bv;
#pragma unroll
      for (int q = 0; q < 4; ++q) bv[q] = (__bf16)aReg[it][q];
      *(bf16x4*)(&Alds[buf][arow[it] * AP + akc[it]]) = bv;
    }
  };
  // Async DMA: 16B per lane, global -> LDS, tracked by ASYNCcnt.
  auto loadB_async = [&](int kt, int buf) {
#pragma unroll
    for (int it = 0; it < 2; ++it) {
      const __bf16* g = W + (long)(kt + brow[it]) * CH + colBlock + bcc[it];
      unsigned ldsoff = (unsigned)(uintptr_t)(&Blds[buf][brow[it] * BP + bcc[it]]);
      unsigned long long ga = (unsigned long long)(uintptr_t)g;
      asm volatile("global_load_async_to_lds_b128 %0, %1, off"
                   :: "v"(ldsoff), "v"(ga) : "memory");
    }
  };

  const int m16 = lane & 15;
  const int khalf = (lane >> 4) << 3;   // 0 or 8

  auto compute = [&](int buf) {
    bf16x16 afrag[2];
#pragma unroll
    for (int i = 0; i < 2; ++i) {
      const __bf16* base = &Alds[buf][(wm * 32 + i * 16 + m16) * AP];
      bf16x8 lo = *(const bf16x8*)(base + khalf);
      bf16x8 hi = *(const bf16x8*)(base + khalf + 16);
      afrag[i] = __builtin_shufflevector(lo, hi, 0, 1, 2, 3, 4, 5, 6, 7,
                                         8, 9, 10, 11, 12, 13, 14, 15);
    }
    bf16x16 bfrag[4];
#pragma unroll
    for (int j = 0; j < 4; ++j) {
      const __bf16* base = &Blds[buf][wn * 64 + j * 16 + m16];
#pragma unroll
      for (int s = 0; s < 8; ++s) {
        bfrag[j][s] = base[(khalf + s) * BP];
        bfrag[j][s + 8] = base[(khalf + 16 + s) * BP];
      }
    }
#pragma unroll
    for (int i = 0; i < 2; ++i)
#pragma unroll
      for (int j = 0; j < 4; ++j)
        acc[i][j] = __builtin_amdgcn_wmma_f32_16x16x32_bf16(
            false, afrag[i], false, bfrag[j], (short)0, acc[i][j], false, false);
  };

  // ---- software pipeline: prologue ----
  loadA(0);
  loadB_async(0, 0);
  storeA(0);
  asm volatile("s_wait_asynccnt 0" ::: "memory");
  __syncthreads();

  int buf = 0;
  for (int kt = 32; kt < K; kt += 32) {
    loadA(kt);                 // global->VGPR for next tile
    loadB_async(kt, buf ^ 1);  // async DMA next weight tile into other buffer
    compute(buf);              // ds_load + wmma from current buffer
    storeA(buf ^ 1);           // VGPR->LDS (other buffer)
    asm volatile("s_wait_asynccnt 0" ::: "memory");
    __syncthreads();
    buf ^= 1;
  }
  compute(buf);

  // ---- epilogue: bias + optional relu, fp32 store ----
  const int nOff = lane & 15;
  const int mHalf = (lane >> 4) << 3;   // 0 or 8
#pragma unroll
  for (int i = 0; i < 2; ++i) {
#pragma unroll
    for (int j = 0; j < 4; ++j) {
      int col = colBlock + wn * 64 + j * 16 + nOff;
      float bv = bias[col];
#pragma unroll
      for (int r = 0; r < 8; ++r) {
        int row = rowBlock + wm * 32 + i * 16 + mHalf + r;
        if (row < M) {
          float v = acc[i][j][r] + bv;
          if (relu) v = fmaxf(v, 0.f);
          C[(long)row * ldc + col] = v;
        }
      }
    }
  }
}

// ---------------------------------------------------------------------------
// Host orchestration
// ---------------------------------------------------------------------------
static inline size_t align256(size_t x) { return (x + 255) & ~(size_t)255; }

extern "C" void kernel_launch(void* const* d_in, const int* in_sizes, int n_in,
                              void* d_out, int out_size, void* d_ws, size_t ws_size,
                              hipStream_t stream) {
  (void)in_sizes; (void)n_in; (void)out_size; (void)ws_size;

  const float* x      = (const float*)d_in[0];
  const int*   eidx   = (const int*)d_in[1];
  const int*   src    = eidx;
  const int*   dst    = eidx + CE;
  const int*   batch  = (const int*)d_in[2];
  const float* target = (const float*)d_in[3];
  const float* w0a = (const float*)d_in[4];  const float* b0a = (const float*)d_in[5];
  const float* w0b = (const float*)d_in[6];  const float* b0b = (const float*)d_in[7];
  const float* w1a = (const float*)d_in[8];  const float* b1a = (const float*)d_in[9];
  const float* w1b = (const float*)d_in[10]; const float* b1b = (const float*)d_in[11];
  const float* w2a = (const float*)d_in[12]; const float* b2a = (const float*)d_in[13];
  const float* w2b = (const float*)d_in[14]; const float* b2b = (const float*)d_in[15];
  const float* bn_gamma = (const float*)d_in[16];
  const float* bn_beta  = (const float*)d_in[17];
  const float* wt  = (const float*)d_in[18]; const float* bt  = (const float*)d_in[19];
  const float* wr1 = (const float*)d_in[20]; const float* br1 = (const float*)d_in[21];
  const float* wr2 = (const float*)d_in[22]; const float* br2 = (const float*)d_in[23];
  float* out = (float*)d_out;

  // ---- workspace carve ----
  const size_t BIG = (size_t)CN * CH * sizeof(float);   // 204.8 MB
  char* cur = (char*)d_ws;
  float* agg  = (float*)cur; cur += align256(BIG);      // also reused as z2
  float* z1   = (float*)cur; cur += align256(BIG);
  float* hA   = (float*)cur; cur += align256(BIG);
  float* hB   = (float*)cur; cur += align256(BIG);
  float* comb = (float*)cur; cur += align256((size_t)CG * 512 * sizeof(float));
  float* r1   = (float*)cur; cur += align256((size_t)CG * CH * sizeof(float));
  float* csum = (float*)cur; cur += align256(512 * sizeof(float));
  float* csq  = csum + 256;
  __bf16* wb0a = (__bf16*)cur; cur += align256((size_t)CF * CH * 2);
  __bf16* wb0b = (__bf16*)cur; cur += align256((size_t)CH * CH * 2);
  __bf16* wb1a = (__bf16*)cur; cur += align256((size_t)CH * CH * 2);
  __bf16* wb1b = (__bf16*)cur; cur += align256((size_t)CH * CH * 2);
  __bf16* wb2a = (__bf16*)cur; cur += align256((size_t)CH * CH * 2);
  __bf16* wb2b = (__bf16*)cur; cur += align256((size_t)CH * CH * 2);
  __bf16* wbt  = (__bf16*)cur; cur += align256((size_t)CT * CH * 2);
  __bf16* wbr1 = (__bf16*)cur; cur += align256((size_t)2 * CH * CH * 2);

  auto conv = [&](const float* s, __bf16* d, int n) {
    k_f32_to_bf16<<<(n + 255) / 256, 256, 0, stream>>>(s, d, n);
  };
  auto zero = [&](float* p, long n) {   // n must be multiple of 4 (it is)
    long n4 = n >> 2;
    k_zero4<<<(unsigned)((n4 + 255) / 256), 256, 0, stream>>>((f32x4*)p, n4);
  };

  // ---- weight conversion (fp32 -> bf16) ----
  conv(w0a, wb0a, CF * CH);
  conv(w0b, wb0b, CH * CH);
  conv(w1a, wb1a, CH * CH);
  conv(w1b, wb1b, CH * CH);
  conv(w2a, wb2a, CH * CH);
  conv(w2b, wb2b, CH * CH);
  conv(wt,  wbt,  CT * CH);
  conv(wr1, wbr1, 2 * CH * CH);

  // ---- one GIN layer ----
  auto layer = [&](const float* hin, int Hin,
                   const __bf16* wa, const float* ba,
                   const __bf16* wb, const float* bb,
                   int l, float* hout) {
    int shift = (Hin == 64) ? 4 : 6;                       // log2(Hin/4)
    zero(agg, (long)CN * Hin);
    {
      long total = (long)CE << shift;
      k_edge_scatter<<<(unsigned)((total + 255) / 256), 256, 0, stream>>>(
          hin, src, dst, agg, CE, Hin, shift);
    }
    dim3 g1((CN + 127) / 128, 2);
    // z1 = relu((hin + agg) @ wa + ba)
    k_wmma_gemm<true><<<g1, 256, 0, stream>>>(hin, agg, wa, ba, z1, CN, Hin, CH, 1);
    // z2 (stored in agg) = z1 @ wb + bb
    k_wmma_gemm<false><<<g1, 256, 0, stream>>>(z1, nullptr, wb, bb, agg, CN, CH, CH, 0);
    // BatchNorm statistics + apply + relu
    zero(csum, 512);
    k_bn_reduce<<<(CN + 511) / 512, 256, 0, stream>>>(agg, csum, csq, CN);
    {
      long total4 = (long)CN * (CH / 4);
      k_bn_apply<<<(unsigned)((total4 + 255) / 256), 256, 0, stream>>>(
          agg, csum, csq, bn_gamma + l * CH, bn_beta + l * CH, hout, CN);
    }
  };

  layer(x,  CF, wb0a, b0a, wb0b, b0b, 0, hA);
  layer(hA, CH, wb1a, b1a, wb1b, b1b, 1, hB);
  layer(hB, CH, wb2a, b2a, wb2b, b2b, 2, hA);

  // ---- graph pooling into combined[:, 0:256] ----
  zero(comb, (long)CG * 512);
  {
    long total = (long)CN * 64;
    k_pool<<<(unsigned)((total + 255) / 256), 256, 0, stream>>>(hA, batch, comb, CN);
  }
  // ---- target projection into combined[:, 256:512] ----
  {
    dim3 g((CG + 127) / 128, 2);
    k_wmma_gemm<false><<<g, 256, 0, stream>>>(target, nullptr, wbt, bt, comb + 256, CG, CT, 512, 1);
  }
  // ---- regressor layer 1: r1 = relu(combined @ wr1 + br1) ----
  {
    dim3 g((CG + 127) / 128, 2);
    k_wmma_gemm<false><<<g, 256, 0, stream>>>(comb, nullptr, wbr1, br1, r1, CG, 2 * CH, CH, 1);
  }
  // ---- final GEMV ----
  k_gemv_out<<<CG / 8, 256, 0, stream>>>(r1, wr2, br2, out, CG);
}
